// blk_encoder_84284438217141
// MI455X (gfx1250) — compile-verified, matrix operand order checked by reading, not run on hardware
//
#include <hip/hip_runtime.h>
#include <hip/hip_bf16.h>

#define BB 256
#define CC 18
#define FF 1024
#define KK 31
#define CF (CC*FF)          // 18432
#define LEPS 1e-5f

typedef __attribute__((ext_vector_type(16))) __bf16         v16bf;
typedef __attribute__((ext_vector_type(16))) unsigned short v16us;
typedef __attribute__((ext_vector_type(8)))  float          v8f;
typedef __attribute__((ext_vector_type(8)))  unsigned int   v8u;

__device__ __forceinline__ unsigned short f2bf(float x) {
  unsigned int u = __float_as_uint(x);
  u += 0x7FFFu + ((u >> 16) & 1u);      // round-to-nearest-even
  return (unsigned short)(u >> 16);
}

__device__ __forceinline__ float gelu_f(float x) {
  return 0.5f * x * (1.0f + erff(x * 0.70710678118654752f));
}

// Build a 16-bit WMMA A/B fragment from LDS rows where K-pairs are contiguous
// (pair r covers K = k0[r], k0[r]+1) -> one aligned b32 LDS load per pair.
__device__ __forceinline__ v16bf frag_u32(const unsigned short* base, const int* k0) {
  v8u t;
#pragma unroll
  for (int r = 0; r < 8; ++r)
    t[r] = *(const unsigned int*)(base + k0[r]);
  return __builtin_bit_cast(v16bf, t);
}

// ---- CDNA5 async global->LDS (16B per lane, ASYNCcnt-tracked) ----
__device__ __forceinline__ unsigned lds_off(const void* p) {
  // LDS aperture: low 32 bits of the flat address are the LDS byte offset
  return (unsigned)(unsigned long long)p;
}
__device__ __forceinline__ void async_ld16(unsigned lds_byte, const void* sbase, unsigned voff) {
  asm volatile("global_load_async_to_lds_b128 %0, %1, %2"
               :: "v"(lds_byte), "v"(voff), "s"(sbase) : "memory");
}
__device__ __forceinline__ void wait_async_le4() {
  asm volatile("s_wait_asynccnt 0x4" ::: "memory");
}
__device__ __forceinline__ void wait_async_0() {
  asm volatile("s_wait_asynccnt 0x0" ::: "memory");
}

// dual block reduction over 256 threads (sum / sumsq)
__device__ __forceinline__ void reduce2(float a, float b, float* red, int tid,
                                        float& A, float& B) {
  __syncthreads();
  red[tid] = a; red[256 + tid] = b;
  __syncthreads();
#pragma unroll
  for (int s = 128; s >= 1; s >>= 1) {
    if (tid < s) { red[tid] += red[tid + s]; red[256 + tid] += red[256 + tid + s]; }
    __syncthreads();
  }
  A = red[0]; B = red[256];
}

// ---------------------------------------------------------------------------
// Kernel 0: mlp_w f32 [c][f][g] -> bf16 TRANSPOSED [c][g][f]
// 64x64 tile transpose per block (grid 16x16x18)
// ---------------------------------------------------------------------------
__global__ __launch_bounds__(256) void wcvt_t(const float* __restrict__ w,
                                              unsigned short* __restrict__ o) {
  __shared__ unsigned short t[64][65];
  const int c  = blockIdx.z;
  const int fb = blockIdx.y * 64;
  const int gb = blockIdx.x * 64;
  const float* src = w + ((size_t)c << 20);
  unsigned short* dst = o + ((size_t)c << 20);
  const int tx = threadIdx.x & 15;       // 16 threads x float4 = 64 cols
  const int ty = threadIdx.x >> 4;       // 16 rows per pass
#pragma unroll
  for (int rr = 0; rr < 64; rr += 16) {
    float4 v = *(const float4*)(src + (size_t)(fb + ty + rr) * FF + gb + 4 * tx);
    t[4 * tx + 0][ty + rr] = f2bf(v.x);
    t[4 * tx + 1][ty + rr] = f2bf(v.y);
    t[4 * tx + 2][ty + rr] = f2bf(v.z);
    t[4 * tx + 3][ty + rr] = f2bf(v.w);
  }
  __syncthreads();
#pragma unroll
  for (int rr = 0; rr < 64; rr += 16) {
    int g = ty + rr;
    ushort4 u;
    u.x = t[g][4 * tx + 0]; u.y = t[g][4 * tx + 1];
    u.z = t[g][4 * tx + 2]; u.w = t[g][4 * tx + 3];
    *(ushort4*)(dst + (size_t)(gb + g) * FF + fb + 4 * tx) = u;
  }
}

// ---------------------------------------------------------------------------
// Kernel 1: per-batch fused  LN1 -> conv(K=31, via im2col WMMA) -> +bias
//           -> LN2 -> GELU -> +x0 (=x1) -> LN3 -> bf16 activations
// one workgroup (256 thr / 8 waves) per batch sample
// ---------------------------------------------------------------------------
__global__ __launch_bounds__(256) void fused_pre(
    const float* __restrict__ x0g, const float* __restrict__ g1, const float* __restrict__ b1,
    const float* __restrict__ cw,  const float* __restrict__ cb,
    const float* __restrict__ g2,  const float* __restrict__ b2,
    const float* __restrict__ g3,  const float* __restrict__ b3,
    float* __restrict__ x1buf, unsigned short* __restrict__ abuf) {
  __shared__ __align__(16) float          xs[CF];          // x0 sample (f32, residual)
  __shared__ __align__(4)  unsigned short lnb[CC * 1056];  // LN1 out bf16, halo 16 each side
  __shared__ __align__(4)  unsigned short wA[32 * 576];    // conv weights im2col [co][ci*32+k]
  __shared__ float red[512];

  const int tid  = threadIdx.x;
  const int bidx = blockIdx.x;
  const int wv   = tid >> 5;
  const int lane = tid & 31;
  const int lm   = lane & 15;
  const int lh   = lane >> 4;

  // ---- load x0 into LDS + LN1 stats ----
  const float4* xg = (const float4*)(x0g + (size_t)bidx * CF);
  float4* xl = (float4*)xs;
  float s1 = 0.f, s2 = 0.f;
  for (int i = tid; i < CF / 4; i += 256) {
    float4 v = xg[i];
    xl[i] = v;
    s1 += v.x + v.y + v.z + v.w;
    s2 += v.x * v.x + v.y * v.y + v.z * v.z + v.w * v.w;
  }
  float S1, S2;
  reduce2(s1, s2, red, tid, S1, S2);
  const float inv_n = 1.0f / (float)CF;
  float mu1 = S1 * inv_n;
  float rs1 = rsqrtf(fmaxf(S2 * inv_n - mu1 * mu1, 0.f) + LEPS);

  // ---- LN1 -> lnb (bf16 with zero halos) ----
  for (int i = tid; i < CF; i += 256) {
    int c = i >> 10, f = i & 1023;
    lnb[c * 1056 + 16 + f] = f2bf((xs[i] - mu1) * rs1 * g1[i] + b1[i]);
  }
  for (int i = tid; i < CC * 32; i += 256) {
    int c = i >> 5, j = i & 31;
    lnb[c * 1056 + ((j < 16) ? j : 1024 + j)] = 0;
  }
  // ---- conv weights -> wA bf16 (K padded 31->32; co padded 18->32, zeros) ----
  for (int i = tid; i < 32 * 576; i += 256) {
    int co = i / 576, kk = i % 576, ci = kk >> 5, k = kk & 31;
    float w = (co < CC && k < KK) ? cw[(co * CC + ci) * KK + k] : 0.f;
    wA[i] = f2bf(w);
  }
  __syncthreads();

  // per-lane K-pair bases for the 16-bit 16x32 WMMA operand layout
  int k0[8];
#pragma unroll
  for (int r = 0; r < 8; ++r)
    k0[r] = ((r < 4) ? (lh ? 8 : 0) : (lh ? 24 : 16)) + 2 * (r & 3);

  v8f acc[2][8];
#pragma unroll
  for (int mt = 0; mt < 2; ++mt)
#pragma unroll
    for (int t = 0; t < 8; ++t) acc[mt][t] = v8f{};

  // ---- conv as GEMM: out[co,f] = sum_{ci,k} wA[co][ci*32+k] * lnb[ci][f+k-15]
  // wave wv owns f-tiles [wv*8, wv*8+8)
  const int tbase = wv * 8;
  for (int s = 0; s < CC; ++s) {              // K-step s == ci
    v16bf Af0 = frag_u32(&wA[(lm)*576 + 32 * s], k0);
    v16bf Af1 = frag_u32(&wA[(lm + 16) * 576 + 32 * s], k0);
    const unsigned short* brow = &lnb[s * 1056 + 1];   // +1 == halo16 - 15
#pragma unroll
    for (int t = 0; t < 8; ++t) {
      const unsigned short* bp = brow + (tbase + t) * 16 + lm;
      v16us bu;
#pragma unroll
      for (int r = 0; r < 8; ++r) {
        bu[2 * r]     = bp[k0[r]];
        bu[2 * r + 1] = bp[k0[r] + 1];
      }
      v16bf Bf = __builtin_bit_cast(v16bf, bu);
      acc[0][t] = __builtin_amdgcn_wmma_f32_16x16x32_bf16(false, Af0, false, Bf,
                                                          (short)0, acc[0][t], false, false);
      acc[1][t] = __builtin_amdgcn_wmma_f32_16x16x32_bf16(false, Af1, false, Bf,
                                                          (short)0, acc[1][t], false, false);
    }
  }

  // ---- + conv bias, LN2 stats ----
  s1 = 0.f; s2 = 0.f;
#pragma unroll
  for (int mt = 0; mt < 2; ++mt)
#pragma unroll
    for (int v = 0; v < 8; ++v) {
      int co = 16 * mt + v + 8 * lh;
      float bias = (co < CC) ? cb[co] : 0.f;
#pragma unroll
      for (int t = 0; t < 8; ++t) {
        float x = acc[mt][t][v] + bias;
        acc[mt][t][v] = x;
        if (co < CC) { s1 += x; s2 += x * x; }
      }
    }
  reduce2(s1, s2, red, tid, S1, S2);
  float mu2 = S1 * inv_n;
  float rs2 = rsqrtf(fmaxf(S2 * inv_n - mu2 * mu2, 0.f) + LEPS);

  // ---- LN2 + GELU + residual -> x1 (in regs), LN3 stats ----
  s1 = 0.f; s2 = 0.f;
#pragma unroll
  for (int mt = 0; mt < 2; ++mt)
#pragma unroll
    for (int v = 0; v < 8; ++v) {
      int co = 16 * mt + v + 8 * lh;
#pragma unroll
      for (int t = 0; t < 8; ++t) {
        int f = (tbase + t) * 16 + lm;
        if (co < CC) {
          int idx = co * FF + f;
          float y  = (acc[mt][t][v] - mu2) * rs2 * g2[idx] + b2[idx];
          float x1 = xs[idx] + gelu_f(y);
          acc[mt][t][v] = x1;
          s1 += x1; s2 += x1 * x1;
        } else {
          acc[mt][t][v] = 0.f;
        }
      }
    }
  reduce2(s1, s2, red, tid, S1, S2);
  float mu3 = S1 * inv_n;
  float rs3 = rsqrtf(fmaxf(S2 * inv_n - mu3 * mu3, 0.f) + LEPS);

  // ---- store x1 (f32, [b][c][f]) and LN3 activations (bf16, [c][b][f]) ----
#pragma unroll
  for (int mt = 0; mt < 2; ++mt)
#pragma unroll
    for (int v = 0; v < 8; ++v) {
      int co = 16 * mt + v + 8 * lh;
      if (co >= CC) continue;
#pragma unroll
      for (int t = 0; t < 8; ++t) {
        int f   = (tbase + t) * 16 + lm;
        int idx = co * FF + f;
        float x1v = acc[mt][t][v];
        x1buf[(size_t)bidx * CF + idx] = x1v;
        float a = (x1v - mu3) * rs3 * g3[idx] + b3[idx];
        abuf[((size_t)co * BB + bidx) * FF + f] = f2bf(a);
      }
    }
}

// ---------------------------------------------------------------------------
// Kernel 2: per-channel GEMM  zz = a @ w  (M=256,N=1024,K=1024 bf16->f32)
// async-to-LDS double-buffered staging; fused bias + GELU + residual store
// grid (ntile=8, mtile=2, c=18), 256 thr / 8 waves, 128x128 tile
// ---------------------------------------------------------------------------
#define LSTR 40   // LDS row stride (ushorts): 80B -> 16B-aligned async stores,
                  // conflict-free frag gathers (20*m mod 64 is a permutation)

__global__ __launch_bounds__(256) void mlp_gemm(
    const unsigned short* __restrict__ abuf, const unsigned short* __restrict__ wbfT,
    const float* __restrict__ mlpb, const float* __restrict__ x1buf,
    float* __restrict__ out) {
  __shared__ __align__(16) unsigned short As[2][128 * LSTR];  // [m][k]
  __shared__ __align__(16) unsigned short Bs[2][128 * LSTR];  // [n][k] (weights pre-transposed)

  const int tid  = threadIdx.x;
  const int c    = blockIdx.z;
  const int m0   = blockIdx.y * 128;
  const int n0   = blockIdx.x * 128;
  const int wv   = tid >> 5;
  const int lane = tid & 31;
  const int lm   = lane & 15;
  const int lh   = lane >> 4;
  const int mw   = 32 * (wv & 3);
  const int nw   = 64 * (wv >> 2);

  int k0[8];
#pragma unroll
  for (int r = 0; r < 8; ++r)
    k0[r] = ((r < 4) ? (lh ? 8 : 0) : (lh ? 24 : 16)) + 2 * (r & 3);

  v8f acc[2][4];
#pragma unroll
  for (int mt = 0; mt < 2; ++mt)
#pragma unroll
    for (int nt = 0; nt < 4; ++nt) acc[mt][nt] = v8f{};

  const unsigned short* Ag = abuf + ((size_t)c * BB + m0) * FF;   // rows: batch
  const unsigned short* Bg = wbfT + ((size_t)c * FF + n0) * FF;   // rows: g (pre-transposed)

  // staging role: thread covers row r=tid>>1, k-half h (16 ushorts = 2x16B)
  const int sr = tid >> 1, sh = (tid & 1) * 16;
  const unsigned aoff0 = (unsigned)((sr * FF + sh) * 2);          // + kb*2 at issue
  const unsigned lrel  = (unsigned)(sr * (LSTR * 2) + sh * 2);    // LDS byte offset in tile
  const unsigned asb0 = lds_off(&As[0][0]), asb1 = lds_off(&As[1][0]);
  const unsigned bsb0 = lds_off(&Bs[0][0]), bsb1 = lds_off(&Bs[1][0]);

  // prologue: stage K-step 0 into buffer 0
  {
    unsigned go = aoff0;
    async_ld16(asb0 + lrel,      Ag, go);
    async_ld16(asb0 + lrel + 16, Ag, go + 16);
    async_ld16(bsb0 + lrel,      Bg, go);
    async_ld16(bsb0 + lrel + 16, Bg, go + 16);
  }

  for (int ks = 0; ks < FF / 32; ++ks) {
    const int cur = ks & 1;
    if (ks + 1 < FF / 32) {
      // prefetch next K-step into the other buffer (safe: trailing barrier of
      // the previous iteration guarantees its consumers are done)
      unsigned go = aoff0 + (unsigned)((ks + 1) * 32 * 2);
      unsigned ab = cur ? asb0 : asb1, bb2 = cur ? bsb0 : bsb1;
      async_ld16(ab + lrel,       Ag, go);
      async_ld16(ab + lrel + 16,  Ag, go + 16);
      async_ld16(bb2 + lrel,      Bg, go);
      async_ld16(bb2 + lrel + 16, Bg, go + 16);
      wait_async_le4();   // in-order completion: current stage is done
    } else {
      wait_async_0();
    }
    __syncthreads();

    const unsigned short* Ab = &As[cur][0];
    const unsigned short* Bb = &Bs[cur][0];
    v16bf Af[2], Bf[4];
#pragma unroll
    for (int mt = 0; mt < 2; ++mt) Af[mt] = frag_u32(Ab + (mw + 16 * mt + lm) * LSTR, k0);
#pragma unroll
    for (int nt = 0; nt < 4; ++nt) Bf[nt] = frag_u32(Bb + (nw + 16 * nt + lm) * LSTR, k0);
#pragma unroll
    for (int mt = 0; mt < 2; ++mt)
#pragma unroll
      for (int nt = 0; nt < 4; ++nt)
        acc[mt][nt] = __builtin_amdgcn_wmma_f32_16x16x32_bf16(
            false, Af[mt], false, Bf[nt], (short)0, acc[mt][nt], false, false);
    __syncthreads();
  }

  // ---- epilogue: + mlp_b, GELU, + x1 residual ----
#pragma unroll
  for (int mt = 0; mt < 2; ++mt)
#pragma unroll
    for (int nt = 0; nt < 4; ++nt)
#pragma unroll
      for (int v = 0; v < 8; ++v) {
        int bb = m0 + mw + 16 * mt + v + 8 * lh;
        int gg = n0 + nw + 16 * nt + lm;
        float zz = acc[mt][nt][v] + mlpb[c * FF + gg];
        size_t o = ((size_t)bb * CC + c) * FF + gg;
        out[o] = x1buf[o] + gelu_f(zz);
      }
}

// ---------------------------------------------------------------------------
extern "C" void kernel_launch(void* const* d_in, const int* in_sizes, int n_in,
                              void* d_out, int out_size, void* d_ws, size_t ws_size,
                              hipStream_t stream) {
  (void)in_sizes; (void)n_in; (void)out_size; (void)ws_size;
  const float* x0 = (const float*)d_in[0];
  const float* g1 = (const float*)d_in[1];
  const float* b1 = (const float*)d_in[2];
  const float* cw = (const float*)d_in[3];
  const float* cb = (const float*)d_in[4];
  const float* g2 = (const float*)d_in[5];
  const float* b2 = (const float*)d_in[6];
  const float* g3 = (const float*)d_in[7];
  const float* b3 = (const float*)d_in[8];
  const float* mw = (const float*)d_in[9];
  const float* mb = (const float*)d_in[10];

  char* ws = (char*)d_ws;
  float* x1buf = (float*)ws;                                        // B*C*F f32
  unsigned short* abuf = (unsigned short*)(ws + (size_t)BB * CF * 4);         // C*B*F bf16
  unsigned short* wbfT = (unsigned short*)(ws + (size_t)BB * CF * 4
                                              + (size_t)CC * BB * FF * 2);    // C*G*F bf16

  wcvt_t<<<dim3(16, 16, CC), dim3(256), 0, stream>>>(mw, wbfT);
  fused_pre<<<dim3(BB), dim3(256), 0, stream>>>(x0, g1, b1, cw, cb, g2, b2, g3, b3,
                                                x1buf, abuf);
  mlp_gemm<<<dim3(8, 2, CC), dim3(256), 0, stream>>>(abuf, wbfT, mb, x1buf, (float*)d_out);
}
